// GENCross_14087492730944
// MI455X (gfx1250) — matrix operand order repaired and checked by reading, not compiled, not run on hardware
//
#include <hip/hip_runtime.h>
#include <stdint.h>

typedef __bf16 bf16;
typedef __attribute__((ext_vector_type(16))) __bf16 bf16x16;
typedef __attribute__((ext_vector_type(8)))  float  f32x8;

#define WMMA_BF16(a,b,c) __builtin_amdgcn_wmma_f32_16x16x32_bf16(false,(a),false,(b),(short)0,(c),false,false)

// ---------------------------------------------------------------------------
// Generic WMMA GEMM:  C(M x Nout) = concat(A0,A1)(M x Kpad) @ Wt^T + bias
//   A rows optionally gathered through idx0/idx1 (graph gather).
//   Wt stored transposed bf16: Wt[n*ldw + k]  (n < Npad, k < Kpad)
//   Outputs: f32 (opt. relu / residual), bf16 row-major (pitch+col offset),
//            bf16 batch-transposed (for building K-major operand tensors).
// Block = 128 threads (4 waves). Block tile: 16 rows x 128 cols.
// A tile (16x32 bf16) double-buffered in LDS via CDNA5 async Global->LDS
// (GLOBAL_LOAD_ASYNC_TO_LDS_B64, ASYNCcnt), shared by all 4 waves.
// ---------------------------------------------------------------------------
struct GemmArgs {
  const bf16* A0; int lda0; const int* idx0;
  const bf16* A1; int lda1; const int* idx1;
  int ksplit;                 // element col where A1 region starts (==Kpad if unused)
  int M, Kpad;
  const bf16* Wt; int ldw; const float* bias;
  int Npad, Nout;
  float* outF; int ldoF; const float* resid; int relu;
  bf16* outB; int ldoB; int colofsB;
  bf16* outT; int MbT;        // outT[((m/MbT)*Nout + n)*MbT + m%MbT]
};

__global__ __launch_bounds__(128)
void gemm_wmma(GemmArgs g) {
  __shared__ alignas(16) bf16 Ad[2][16 * 32];
  const int m0   = blockIdx.x * 16;
  const int wave = threadIdx.x >> 5;
  const int lane = threadIdx.x & 31;
  const int n0   = blockIdx.y * 128 + wave * 32;
  const bool low = lane < 16;
  const int  r   = lane & 15;

  // per-thread staging assignment (K-invariant, hoisted out of the loop)
  const int t     = threadIdx.x;
  const int row   = t >> 3;      // 0..15
  const int chunk = t & 7;       // 8B chunk within the 64B row
  const int grow  = m0 + row;
  const int arow0 = g.idx0 ? g.idx0[grow] : grow;
  const int arow1 = g.idx1 ? g.idx1[grow] : grow;
  const bf16* rowp0 = g.A0 + (size_t)arow0 * g.lda0;
  const bf16* rowp1 = g.A1 + (size_t)arow1 * g.lda1 - g.ksplit;
  const unsigned lds0 = (unsigned)(uintptr_t)(&Ad[0][0]) + (unsigned)t * 8u;
  const unsigned lds1 = (unsigned)(uintptr_t)(&Ad[1][0]) + (unsigned)t * 8u;

  const int ktiles = g.Kpad >> 5;
  auto stage = [&](int kt) {
    const int kc = (kt << 5) + (chunk << 2);
    const bf16* sp = (kc >= g.ksplit ? rowp1 : rowp0) + kc;
    const unsigned ld = (kt & 1) ? lds1 : lds0;
    // CDNA5 async Global->LDS copy (8B per lane), tracked by ASYNCcnt
    asm volatile("global_load_async_to_lds_b64 %0, %1, off"
                 :: "v"(ld), "v"(sp) : "memory");
  };

  f32x8 acc0 = {};
  f32x8 acc1 = {};
  stage(0);
  for (int kt = 0; kt < ktiles; ++kt) {
    asm volatile("s_wait_asynccnt 0x0" ::: "memory");
    __syncthreads();
    if (kt + 1 < ktiles) stage(kt + 1); // overlap next tile with this compute

    const bf16* At = Ad[kt & 1];
    bf16x16 af; // A fragment: lo lanes K[0..7],[16..23]; hi lanes K[8..15],[24..31]
    {
      const uint4* lp = (const uint4*)(At + r * 32 + (low ? 0 : 8));
      ((uint4*)&af)[0] = lp[0];
      ((uint4*)&af)[1] = lp[2];
    }

    const int kbase = (kt << 5) + (low ? 0 : 8);
    int nr0 = n0 + r;      if (nr0 > g.Npad - 1) nr0 = g.Npad - 1;
    int nr1 = n0 + 16 + r; if (nr1 > g.Npad - 1) nr1 = g.Npad - 1;
    bf16x16 bf0, bf1; // B fragment: lane<->N, lo lanes K[0..7],[16..23]; hi lanes K[8..15],[24..31]
    {
      const uint4* w0 = (const uint4*)(g.Wt + (size_t)nr0 * g.ldw + kbase);
      ((uint4*)&bf0)[0] = w0[0]; ((uint4*)&bf0)[1] = w0[2];
      if (kt + 1 < ktiles) __builtin_prefetch((const char*)w0 + 64, 0, 1);
      const uint4* w1 = (const uint4*)(g.Wt + (size_t)nr1 * g.ldw + kbase);
      ((uint4*)&bf1)[0] = w1[0]; ((uint4*)&bf1)[1] = w1[2];
    }

    acc0 = WMMA_BF16(af, bf0, acc0);
    acc1 = WMMA_BF16(af, bf1, acc1);
  }

#pragma unroll
  for (int tt = 0; tt < 2; ++tt) {
    f32x8 acc = tt ? acc1 : acc0;
    const int col = n0 + 16 * tt + r;
    if (col >= g.Nout) continue;
    const float bv = g.bias ? g.bias[col] : 0.f;
#pragma unroll
    for (int e = 0; e < 8; ++e) {
      const int m = m0 + e + (low ? 0 : 8);
      float v = acc[e] + bv;
      if (g.relu)  v = v > 0.f ? v : 0.f;
      if (g.resid) v += g.resid[(size_t)m * g.ldoF + col];
      if (g.outF)  g.outF[(size_t)m * g.ldoF + col] = v;
      if (g.outB)  g.outB[(size_t)m * g.ldoB + g.colofsB + col] = (bf16)v;
      if (g.outT) {
        const int bb = m / g.MbT, mm = m % g.MbT;
        g.outT[((size_t)bb * g.Nout + col) * g.MbT + mm] = (bf16)v;
      }
    }
  }
}

// ---------------------------------------------------------------------------
// softmax(-d2) over 4096 latent positions; writes scores^T (4096 x 1024) bf16.
// -d2 = 2*xc.pos - |pos|^2 (row-constant |xc|^2 cancels in softmax).
// One wave per context point.
// ---------------------------------------------------------------------------
__global__ __launch_bounds__(128)
void scores_softmax(const float* __restrict__ xc, const float* __restrict__ pos,
                    bf16* __restrict__ scoresT, int b) {
  const int lane = threadIdx.x & 31;
  const int n = blockIdx.x * 4 + (threadIdx.x >> 5);
  const float x0 = xc[((size_t)b * 1024 + n) * 2 + 0];
  const float x1 = xc[((size_t)b * 1024 + n) * 2 + 1];
  float mx = -3.0e38f;
  for (int m = lane; m < 4096; m += 32) {
    float p0 = pos[m * 2], p1 = pos[m * 2 + 1];
    float s = 2.f * (x0 * p0 + x1 * p1) - (p0 * p0 + p1 * p1);
    mx = s > mx ? s : mx;
  }
#pragma unroll
  for (int o = 16; o; o >>= 1) { float t = __shfl_xor(mx, o, 32); mx = t > mx ? t : mx; }
  float sum = 0.f;
  for (int m = lane; m < 4096; m += 32) {
    float p0 = pos[m * 2], p1 = pos[m * 2 + 1];
    float s = 2.f * (x0 * p0 + x1 * p1) - (p0 * p0 + p1 * p1);
    sum += __expf(s - mx);
  }
#pragma unroll
  for (int o = 16; o; o >>= 1) sum += __shfl_xor(sum, o, 32);
  const float inv = 1.f / sum;
  for (int m = lane; m < 4096; m += 32) {
    float p0 = pos[m * 2], p1 = pos[m * 2 + 1];
    float s = 2.f * (x0 * p0 + x1 * p1) - (p0 * p0 + p1 * p1);
    scoresT[(size_t)m * 1024 + n] = (bf16)(__expf(s - mx) * inv);
  }
}

// LN over 130 message features, fused with scatter-add into inbox (f32 atomics).
__global__ __launch_bounds__(128)
void ln_scatter(const float* __restrict__ msg, const float* __restrict__ g1,
                const float* __restrict__ b1, const int* __restrict__ recv,
                float* __restrict__ inbox) {
  const int lane = threadIdx.x & 31;
  const int row = blockIdx.x * 4 + (threadIdx.x >> 5);
  const float* x = msg + (size_t)row * 160;
  float s = 0.f, s2 = 0.f;
  for (int c = lane; c < 130; c += 32) { float v = x[c]; s += v; s2 += v * v; }
#pragma unroll
  for (int o = 16; o; o >>= 1) { s += __shfl_xor(s, o, 32); s2 += __shfl_xor(s2, o, 32); }
  const float mu = s * (1.f / 130.f);
  const float var = s2 * (1.f / 130.f) - mu * mu;
  const float rs = rsqrtf(var + 1e-5f);
  float* dst = inbox + (size_t)recv[row] * 160;
  for (int c = lane; c < 130; c += 32) {
    atomicAdd(dst + c, (x[c] - mu) * rs * g1[c] + b1[c]);
  }
}

// LN over 128 node features; writes both node buffer (pitch 160, col 2..) and
// compact latent buffer (pitch 128) in bf16.
__global__ __launch_bounds__(128)
void ln_node(const float* __restrict__ h, const float* __restrict__ g2,
             const float* __restrict__ b2, bf16* __restrict__ nodesb,
             bf16* __restrict__ lat) {
  const int lane = threadIdx.x & 31;
  const int row = blockIdx.x * 4 + (threadIdx.x >> 5);
  const float* x = h + (size_t)row * 128;
  float s = 0.f, s2 = 0.f;
  for (int c = lane; c < 128; c += 32) { float v = x[c]; s += v; s2 += v * v; }
#pragma unroll
  for (int o = 16; o; o >>= 1) { s += __shfl_xor(s, o, 32); s2 += __shfl_xor(s2, o, 32); }
  const float mu = s * (1.f / 128.f);
  const float var = s2 * (1.f / 128.f) - mu * mu;
  const float rs = rsqrtf(var + 1e-5f);
  for (int c = lane; c < 128; c += 32) {
    float v = (x[c] - mu) * rs * g2[c] + b2[c];
    nodesb[(size_t)row * 160 + 2 + c] = (bf16)v;
    lat[(size_t)row * 128 + c] = (bf16)v;
  }
}

// ---------------------------------------------------------------------------
// Flash cross-attention: one wave per (b, head, 16-query tile).
// QK^T: one WMMA per 16-key tile (K = head dim 32). Online softmax with
// width-16 shuffles (rows live per element across a lane half in C layout).
// P tile bounces through per-wave LDS to re-layout C -> A for the PV WMMA.
// V consumed from a dim-major transpose vT (B,128,4096).
// ---------------------------------------------------------------------------
__global__ __launch_bounds__(128)
void attn_flash(const bf16* __restrict__ qh, const bf16* __restrict__ kh,
                const bf16* __restrict__ vT, bf16* __restrict__ oB) {
  __shared__ alignas(16) bf16 sm[4][16 * 32];
  const int lane = threadIdx.x & 31;
  const int wv = threadIdx.x >> 5;
  const int qt = blockIdx.x * 4 + wv;
  const int h = blockIdx.y;
  const int b = blockIdx.z;
  const bool low = lane < 16;
  const int r = lane & 15;
  bf16* smem = sm[wv];

  bf16x16 qf;
  {
    const uint4* qr = (const uint4*)(qh + ((size_t)(b * 1024 + qt * 16 + r)) * 128 + h * 32 + (low ? 0 : 8));
    ((uint4*)&qf)[0] = qr[0];
    ((uint4*)&qf)[1] = qr[2];
  }
  f32x8 accO0 = {};
  f32x8 accO1 = {};
  float rmax[8], rsum[8];
#pragma unroll
  for (int e = 0; e < 8; ++e) { rmax[e] = -3.0e38f; rsum[e] = 0.f; }
  const float scale = 0.17677669529663687f; // 1/sqrt(32)

  for (int k0 = 0; k0 < 4096; k0 += 32) {
    bf16x16 kf0, kf1;
    {
      const uint4* kr0 = (const uint4*)(kh + ((size_t)(b * 4096 + k0 + r)) * 128 + h * 32 + (low ? 0 : 8));
      ((uint4*)&kf0)[0] = kr0[0]; ((uint4*)&kf0)[1] = kr0[2];
      const uint4* kr1 = (const uint4*)(kh + ((size_t)(b * 4096 + k0 + 16 + r)) * 128 + h * 32 + (low ? 0 : 8));
      ((uint4*)&kf1)[0] = kr1[0]; ((uint4*)&kf1)[1] = kr1[2];
    }
    f32x8 s0 = {};
    f32x8 s1 = {};
    s0 = WMMA_BF16(qf, kf0, s0);
    s1 = WMMA_BF16(qf, kf1, s1);
    __syncthreads();
#pragma unroll
    for (int e = 0; e < 8; ++e) {
      float v0 = s0[e] * scale, v1 = s1[e] * scale;
      float mx = v0 > v1 ? v0 : v1;
#pragma unroll
      for (int o = 8; o; o >>= 1) { float t = __shfl_xor(mx, o, 32); mx = t > mx ? t : mx; }
      const float nm = rmax[e] > mx ? rmax[e] : mx;
      const float corr = __expf(rmax[e] - nm);
      rmax[e] = nm;
      const float e0 = __expf(v0 - nm), e1 = __expf(v1 - nm);
      float ps = e0 + e1;
#pragma unroll
      for (int o = 8; o; o >>= 1) ps += __shfl_xor(ps, o, 32);
      rsum[e] = rsum[e] * corr + ps;
      accO0[e] *= corr; accO1[e] *= corr;
      const int rr = e + (low ? 0 : 8);
      smem[rr * 32 + r] = (bf16)e0;
      smem[rr * 32 + 16 + r] = (bf16)e1;
    }
    __syncthreads();
    bf16x16 pf;
    {
      const uint4* pp = (const uint4*)(smem + r * 32 + (low ? 0 : 8));
      ((uint4*)&pf)[0] = pp[0];
      ((uint4*)&pf)[1] = pp[2];
    }
    bf16x16 vf0, vf1;
    {
      const uint4* vv0 = (const uint4*)(vT + ((size_t)b * 128 + h * 32 + r) * 4096 + k0 + (low ? 0 : 8));
      ((uint4*)&vf0)[0] = vv0[0]; ((uint4*)&vf0)[1] = vv0[2];
      const uint4* vv1 = (const uint4*)(vT + ((size_t)b * 128 + h * 32 + 16 + r) * 4096 + k0 + (low ? 0 : 8));
      ((uint4*)&vf1)[0] = vv1[0]; ((uint4*)&vf1)[1] = vv1[2];
    }
    accO0 = WMMA_BF16(pf, vf0, accO0);
    accO1 = WMMA_BF16(pf, vf1, accO1);
  }
#pragma unroll
  for (int e = 0; e < 8; ++e) {
    const float inv = 1.f / rsum[e];
    const int m = qt * 16 + e + (low ? 0 : 8);
    bf16* orow = oB + ((size_t)(b * 1024 + m)) * 128 + h * 32;
    orow[r] = (bf16)(accO0[e] * inv);
    orow[16 + r] = (bf16)(accO1[e] * inv);
  }
}

// ------------------------------ pack / util --------------------------------
__global__ void pack_w(const float* W, int K, int N, bf16* Wt, int Kpad, int Npad, int split) {
  int t = blockIdx.x * 256 + threadIdx.x;
  if (t >= Kpad * Npad) return;
  int n = t / Kpad, k = t % Kpad;
  float v = 0.f;
  if (split < 0) {
    if (k < K && n < N) v = W[(size_t)k * N + n];
  } else {
    const int half = Kpad / 2;
    int srck = -1;
    if (k < split) srck = k;
    else if (k >= half && k < half + (K - split)) srck = k - half + split;
    if (srck >= 0 && n < N) v = W[(size_t)srck * N + n];
  }
  Wt[(size_t)n * Kpad + k] = (bf16)v;
}
__global__ void pack_bias(const float* b, int N, float* dst, int Npad) {
  int t = blockIdx.x * 256 + threadIdx.x;
  if (t < Npad) dst[t] = t < N ? b[t] : 0.f;
}
__global__ void pack_enc_in(const float* xc, const float* yc, bf16* A) {
  int m = blockIdx.x * 256 + threadIdx.x;
  if (m >= 8192) return;
  bf16* rr = A + (size_t)m * 32;
  for (int c = 0; c < 32; ++c) rr[c] = (bf16)0.f;
  rr[0] = (bf16)xc[m * 2]; rr[1] = (bf16)xc[m * 2 + 1]; rr[2] = (bf16)yc[m];
}
__global__ void pack_q_in(const float* xt, bf16* A) {
  int m = blockIdx.x * 256 + threadIdx.x;
  if (m >= 8192) return;
  bf16* rr = A + (size_t)m * 32;
  for (int c = 0; c < 32; ++c) rr[c] = (bf16)0.f;
  rr[0] = (bf16)xt[m * 2]; rr[1] = (bf16)xt[m * 2 + 1];
}
__global__ void pack_nodes(const float* pos, bf16* nodes) {
  int t = blockIdx.x * 256 + threadIdx.x;
  if (t >= 8 * 4096) return;
  const int i = t & 4095;
  bf16* rr = nodes + (size_t)t * 160;
  rr[0] = (bf16)pos[i * 2]; rr[1] = (bf16)pos[i * 2 + 1];
  for (int c = 130; c < 160; ++c) rr[c] = (bf16)0.f;
}
__global__ void zerof(float* p, int n) {
  int t = blockIdx.x * 256 + threadIdx.x;
  if (t < n) p[t] = 0.f;
}
__global__ void f2b(const float* s, bf16* d, int n) {
  int t = blockIdx.x * 256 + threadIdx.x;
  if (t < n) d[t] = (bf16)s[t];
}

// ------------------------------ host side ----------------------------------
static void G(hipStream_t st,
              const bf16* A0, int lda0, const int* idx0,
              const bf16* A1, int lda1, const int* idx1, int ksplit,
              int M, int Kpad,
              const bf16* Wt, int ldw, const float* bias, int Npad, int Nout,
              float* outF, int ldoF, const float* resid, int relu,
              bf16* outB, int ldoB, int colofsB,
              bf16* outT, int MbT) {
  GemmArgs g;
  g.A0 = A0; g.lda0 = lda0; g.idx0 = idx0;
  g.A1 = A1; g.lda1 = lda1; g.idx1 = idx1; g.ksplit = ksplit;
  g.M = M; g.Kpad = Kpad; g.Wt = Wt; g.ldw = ldw; g.bias = bias;
  g.Npad = Npad; g.Nout = Nout;
  g.outF = outF; g.ldoF = ldoF; g.resid = resid; g.relu = relu;
  g.outB = outB; g.ldoB = ldoB; g.colofsB = colofsB;
  g.outT = outT; g.MbT = MbT;
  dim3 grid(M / 16, (Npad + 127) / 128);
  gemm_wmma<<<grid, 128, 0, st>>>(g);
}
static void PW(hipStream_t st, const float* W, int K, int N, bf16* Wt, int Kp, int Np, int split) {
  int tot = Kp * Np;
  pack_w<<<(tot + 255) / 256, 256, 0, st>>>(W, K, N, Wt, Kp, Np, split);
}
static void PB(hipStream_t st, const float* b, int N, float* dst, int Np) {
  pack_bias<<<(Np + 255) / 256, 256, 0, st>>>(b, N, dst, Np);
}

extern "C" void kernel_launch(void* const* d_in, const int* in_sizes, int n_in,
                              void* d_out, int out_size, void* d_ws, size_t ws_size,
                              hipStream_t stream) {
  (void)in_sizes; (void)n_in; (void)out_size; (void)ws_size;
  const float* xc  = (const float*)d_in[0];
  const float* yc  = (const float*)d_in[1];
  const float* xt  = (const float*)d_in[2];
  const float* pos = (const float*)d_in[3];
  const int* senders   = (const int*)d_in[4];
  const int* receivers = (const int*)d_in[5];
  const float *eW0 = (const float*)d_in[6],  *eB0 = (const float*)d_in[7];
  const float *eW1 = (const float*)d_in[8],  *eB1 = (const float*)d_in[9];
  const float *eW2 = (const float*)d_in[10], *eB2 = (const float*)d_in[11];
  const float *qW0 = (const float*)d_in[12], *qB0 = (const float*)d_in[13];
  const float *qW1 = (const float*)d_in[14], *qB1 = (const float*)d_in[15];
  const float *qW2 = (const float*)d_in[16], *qB2 = (const float*)d_in[17];
  const float *dW0 = (const float*)d_in[18], *dB0 = (const float*)d_in[19];
  const float *dW1 = (const float*)d_in[20], *dB1 = (const float*)d_in[21];
  const float *Wm = (const float*)d_in[22], *bm = (const float*)d_in[23];
  const float *g1 = (const float*)d_in[24], *b1 = (const float*)d_in[25];
  const float *Wn = (const float*)d_in[26], *bn = (const float*)d_in[27];
  const float *g2 = (const float*)d_in[28], *b2 = (const float*)d_in[29];
  const float *Wq = (const float*)d_in[30], *bq = (const float*)d_in[31];
  const float *Wk = (const float*)d_in[32], *bk = (const float*)d_in[33];
  const float *Wv = (const float*)d_in[34], *bv = (const float*)d_in[35];
  const float *Wo = (const float*)d_in[36], *bo = (const float*)d_in[37];
  float* out = (float*)d_out;

  uint8_t* base = (uint8_t*)d_ws; size_t off = 0;
  auto A = [&](size_t bytes) -> void* {
    void* p = base + off; off = (off + bytes + 255) & ~(size_t)255; return p;
  };
  // packed weights / biases
  bf16* WtE0 = (bf16*)A(128 * 32 * 2);   float* bE0 = (float*)A(128 * 4);
  bf16* WtE1 = (bf16*)A(128 * 128 * 2);  float* bE1 = (float*)A(128 * 4);
  bf16* WtE2 = (bf16*)A(128 * 128 * 2);  float* bE2 = (float*)A(128 * 4);
  bf16* WtQ0 = (bf16*)A(128 * 32 * 2);   float* bQ0 = (float*)A(128 * 4);
  bf16* WtQ1 = (bf16*)A(128 * 128 * 2);  float* bQ1 = (float*)A(128 * 4);
  bf16* WtQ2 = (bf16*)A(128 * 128 * 2);  float* bQ2 = (float*)A(128 * 4);
  bf16* WtD0 = (bf16*)A(128 * 128 * 2);  float* bDD0 = (float*)A(128 * 4);
  bf16* WtD1 = (bf16*)A(16 * 128 * 2);   float* bDD1 = (float*)A(16 * 4);
  bf16* WtM  = (bf16*)A(160 * 320 * 2);  float* bMM = (float*)A(160 * 4);
  bf16* WtN  = (bf16*)A(128 * 320 * 2);  float* bNN = (float*)A(128 * 4);
  bf16* WtWq = (bf16*)A(128 * 128 * 2);  float* bWq = (float*)A(128 * 4);
  bf16* WtWk = (bf16*)A(128 * 128 * 2);  float* bWk = (float*)A(128 * 4);
  bf16* WtWv = (bf16*)A(128 * 128 * 2);  float* bWv = (float*)A(128 * 4);
  bf16* WtWo = (bf16*)A(128 * 128 * 2);  float* bWo = (float*)A(128 * 4);
  // activations
  bf16* Ae      = (bf16*)A((size_t)8192 * 32 * 2);
  bf16* hA      = (bf16*)A((size_t)8192 * 128 * 2);
  bf16* hB      = (bf16*)A((size_t)8192 * 128 * 2);
  bf16* embT    = (bf16*)A((size_t)8 * 128 * 1024 * 2);
  bf16* scoresT = (bf16*)A((size_t)4096 * 1024 * 2);
  bf16* nodes   = (bf16*)A((size_t)8 * 4096 * 160 * 2);
  bf16* lat     = (bf16*)A((size_t)8 * 4096 * 128 * 2);
  float* msgF   = (float*)A((size_t)32768 * 160 * 4);
  float* inbox  = (float*)A((size_t)4096 * 160 * 4);
  bf16* inboxB  = (bf16*)A((size_t)4096 * 160 * 2);
  float* hF     = (float*)A((size_t)4096 * 128 * 4);
  bf16* Aq      = (bf16*)A((size_t)8192 * 32 * 2);
  float* qF     = (float*)A((size_t)8192 * 128 * 4);
  bf16* qbf     = (bf16*)A((size_t)8192 * 128 * 2);
  bf16* qhB     = (bf16*)A((size_t)8192 * 128 * 2);
  bf16* khB     = (bf16*)A((size_t)32768 * 128 * 2);
  bf16* vT      = (bf16*)A((size_t)8 * 128 * 4096 * 2);
  bf16* oB      = (bf16*)A((size_t)8192 * 128 * 2);
  bf16* zbf     = (bf16*)A((size_t)8192 * 128 * 2);
  bf16* dh      = (bf16*)A((size_t)8192 * 128 * 2);

  // --- pack weights (bf16, transposed, padded) ---
  PW(stream, eW0, 3, 128, WtE0, 32, 128, -1);   PB(stream, eB0, 128, bE0, 128);
  PW(stream, eW1, 128, 128, WtE1, 128, 128, -1); PB(stream, eB1, 128, bE1, 128);
  PW(stream, eW2, 128, 128, WtE2, 128, 128, -1); PB(stream, eB2, 128, bE2, 128);
  PW(stream, qW0, 2, 128, WtQ0, 32, 128, -1);    PB(stream, qB0, 128, bQ0, 128);
  PW(stream, qW1, 128, 128, WtQ1, 128, 128, -1); PB(stream, qB1, 128, bQ1, 128);
  PW(stream, qW2, 128, 128, WtQ2, 128, 128, -1); PB(stream, qB2, 128, bQ2, 128);
  PW(stream, dW0, 128, 128, WtD0, 128, 128, -1); PB(stream, dB0, 128, bDD0, 128);
  PW(stream, dW1, 128, 1, WtD1, 128, 16, -1);    PB(stream, dB1, 1, bDD1, 16);
  PW(stream, Wm, 260, 130, WtM, 320, 160, 130);  PB(stream, bm, 130, bMM, 160);
  PW(stream, Wn, 260, 128, WtN, 320, 128, 130);  PB(stream, bn, 128, bNN, 128);
  PW(stream, Wq, 128, 128, WtWq, 128, 128, -1);  PB(stream, bq, 128, bWq, 128);
  PW(stream, Wk, 128, 128, WtWk, 128, 128, -1);  PB(stream, bk, 128, bWk, 128);
  PW(stream, Wv, 128, 128, WtWv, 128, 128, -1);  PB(stream, bv, 128, bWv, 128);
  PW(stream, Wo, 128, 128, WtWo, 128, 128, -1);  PB(stream, bo, 128, bWo, 128);

  pack_enc_in<<<32, 256, 0, stream>>>(xc, yc, Ae);
  pack_q_in<<<32, 256, 0, stream>>>(xt, Aq);
  pack_nodes<<<128, 256, 0, stream>>>(pos, nodes);

  // --- encoder MLP: (xc,yc) -> emb; final layer writes batch-transposed embT ---
  G(stream, Ae, 32, nullptr, Ae, 32, nullptr, 32, 8192, 32,
    WtE0, 32, bE0, 128, 128, nullptr, 0, nullptr, 1, hA, 128, 0, nullptr, 1);
  G(stream, hA, 128, nullptr, hA, 128, nullptr, 128, 8192, 128,
    WtE1, 128, bE1, 128, 128, nullptr, 0, nullptr, 1, hB, 128, 0, nullptr, 1);
  G(stream, hB, 128, nullptr, hB, 128, nullptr, 128, 8192, 128,
    WtE2, 128, bE2, 128, 128, nullptr, 0, nullptr, 0, nullptr, 0, 0, embT, 1024);

  // --- scores + latents projection (per batch; K=1024 WMMA GEMM) ---
  for (int b = 0; b < 8; ++b) {
    scores_softmax<<<256, 128, 0, stream>>>(xc, pos, scoresT, b);
    G(stream, scoresT, 1024, nullptr, scoresT, 1024, nullptr, 1024, 4096, 1024,
      embT + (size_t)b * 128 * 1024, 1024, nullptr, 128, 128,
      nullptr, 0, nullptr, 0, nodes + (size_t)b * 4096 * 160, 160, 2, nullptr, 1);
  }

  // --- 8 GN message-passing steps ---
  for (int s = 0; s < 8; ++s) {
    for (int b = 0; b < 8; ++b) {
      bf16* nb = nodes + (size_t)b * 4096 * 160;
      zerof<<<(4096 * 160 + 255) / 256, 256, 0, stream>>>(inbox, 4096 * 160);
      // messages = concat(nodes[recv], nodes[send]) @ Wm + bm   (gathered WMMA GEMM)
      G(stream, nb, 160, receivers, nb, 160, senders, 160, 32768, 320,
        WtM, 320, bMM, 160, 130, msgF, 160, nullptr, 0, nullptr, 0, 0, nullptr, 1);
      ln_scatter<<<8192, 128, 0, stream>>>(msgF, g1, b1, receivers, inbox);
      f2b<<<(4096 * 160 + 255) / 256, 256, 0, stream>>>(inbox, inboxB, 4096 * 160);
      // h = concat(nodes, inbox) @ Wn + bn
      G(stream, nb, 160, nullptr, inboxB, 160, nullptr, 160, 4096, 320,
        WtN, 320, bNN, 128, 128, hF, 128, nullptr, 0, nullptr, 0, 0, nullptr, 1);
      ln_node<<<1024, 128, 0, stream>>>(hF, g2, b2, nb, lat + (size_t)b * 4096 * 128);
    }
  }

  // --- query MLP + attention projections ---
  G(stream, Aq, 32, nullptr, Aq, 32, nullptr, 32, 8192, 32,
    WtQ0, 32, bQ0, 128, 128, nullptr, 0, nullptr, 1, hA, 128, 0, nullptr, 1);
  G(stream, hA, 128, nullptr, hA, 128, nullptr, 128, 8192, 128,
    WtQ1, 128, bQ1, 128, 128, nullptr, 0, nullptr, 1, hB, 128, 0, nullptr, 1);
  G(stream, hB, 128, nullptr, hB, 128, nullptr, 128, 8192, 128,
    WtQ2, 128, bQ2, 128, 128, qF, 128, nullptr, 0, qbf, 128, 0, nullptr, 1);
  G(stream, qbf, 128, nullptr, qbf, 128, nullptr, 128, 8192, 128,
    WtWq, 128, bWq, 128, 128, nullptr, 0, nullptr, 0, qhB, 128, 0, nullptr, 1);
  G(stream, lat, 128, nullptr, lat, 128, nullptr, 128, 32768, 128,
    WtWk, 128, bWk, 128, 128, nullptr, 0, nullptr, 0, khB, 128, 0, nullptr, 1);
  G(stream, lat, 128, nullptr, lat, 128, nullptr, 128, 32768, 128,
    WtWv, 128, bWv, 128, 128, nullptr, 0, nullptr, 0, nullptr, 0, 0, vT, 4096);

  attn_flash<<<dim3(16, 4, 8), 128, 0, stream>>>(qhB, khB, vT, oB);

  // z = q + o @ Wo + bo ; decoder MLP -> out
  G(stream, oB, 128, nullptr, oB, 128, nullptr, 128, 8192, 128,
    WtWo, 128, bWo, 128, 128, nullptr, 128, qF, 0, zbf, 128, 0, nullptr, 1);
  G(stream, zbf, 128, nullptr, zbf, 128, nullptr, 128, 8192, 128,
    WtD0, 128, bDD0, 128, 128, nullptr, 0, nullptr, 1, dh, 128, 0, nullptr, 1);
  G(stream, dh, 128, nullptr, dh, 128, nullptr, 128, 8192, 128,
    WtD1, 128, bDD1, 16, 1, out, 1, nullptr, 0, nullptr, 0, 0, nullptr, 1);
}
